// MFN_58411555225773
// MI455X (gfx1250) — matrix operand order repaired and checked by reading, not compiled, not run on hardware
//
#include <hip/hip_runtime.h>
#include <math.h>

// ---------------- CDNA5 WMMA plumbing ----------------
typedef __attribute__((ext_vector_type(16))) __bf16 v16bf;
typedef __attribute__((ext_vector_type(8)))  float  v8f;

#define DEV __device__ __forceinline__

struct __attribute__((aligned(16))) U4 { unsigned int x, y, z, w; };

union BFrag {
    v16bf v;
    unsigned short u[16];
    U4 q[2];
};

DEV unsigned short f2bfraw(float f) {
    union { float f; unsigned u; } c; c.f = f;
    unsigned u = c.u;
    unsigned r = u + 0x7FFFu + ((u >> 16) & 1u);   // round-to-nearest-even
    return (unsigned short)(r >> 16);
}

DEV v8f vzero() {
    v8f z;
#pragma unroll
    for (int i = 0; i < 8; i++) z[i] = 0.f;
    return z;
}

// A-fragment (16x32 bf16) from row-major bf16 [M,K].
// ISA layout: lanes0-15 row=m0+lane, elems0-7 => K=kk+0..7, elems8-15 => K=kk+16..23
//             lanes16-31 row=m0+lane-16, elems0-7 => K=kk+8..15, elems8-15 => K=kk+24..31
DEV v16bf load_a_bf(const unsigned short* __restrict__ A, int lda, int m0, int kk, int lane) {
    const unsigned short* p = A + (size_t)(m0 + (lane & 15)) * (size_t)lda + kk + ((lane & 16) ? 8 : 0);
    BFrag r;
    r.q[0] = *(const U4*)(p);
    r.q[1] = *(const U4*)(p + 16);
    return r.v;
}

// B-fragment (32x16 bf16), B = W^T with W row-major [Nout, K].
// elem e at lane l => K = kk + (l>=16 ? 8 : 0) + (e<8 ? e : 16 + e - 8), col = n0 + (l&15)
DEV v16bf load_b_bf(const unsigned short* __restrict__ W, int ldw, int n0, int kk, int lane) {
    const unsigned short* p = W + (size_t)(n0 + (lane & 15)) * (size_t)ldw + kk + ((lane & 16) ? 8 : 0);
    BFrag r;
    r.q[0] = *(const U4*)(p);
    r.q[1] = *(const U4*)(p + 16);
    return r.v;
}

DEV v8f wmma_bf16(v16bf a, v16bf b, v8f c) {
    return __builtin_amdgcn_wmma_f32_16x16x32_bf16(false, a, false, b, (short)0, c, false, false);
}

DEV float sigmoidf_(float x) { return 1.0f / (1.0f + expf(-x)); }

// ---------------- constants ----------------
#define TT 128
#define NB 256
#define D_T 768
#define D_A 128
#define D_V 512
#define H_T 256
#define H_A 128
#define H_V 128
#define HD  512
#define MEM 256

// ---------------- fp32 -> bf16 conversion (weights + inputs, once per launch) ----------------
__global__ void k_cvt(const float* __restrict__ src, unsigned short* __restrict__ dst, int n) {
    int i = blockIdx.x * blockDim.x + threadIdx.x;
    if (i < n) dst[i] = f2bfraw(src[i]);
}

// ---------------- generic GEMM: C = act(A[M,K]bf16 @ W[N,K]^T bf16 + bias) ----------------
// mode: 0 = identity -> fp32 out, 1 = relu -> bf16 out, 2 = relu -> fp32 out
// wave computes 16x64; software-pipelined: next iteration's fragments are loaded
// before the current iteration's WMMAs so loads overlap the matrix pipe.
__global__ void k_gemm(const unsigned short* __restrict__ A, const unsigned short* __restrict__ W,
                       const float* __restrict__ bias, float* __restrict__ Cf,
                       unsigned short* __restrict__ Cb,
                       int M, int N, int K, int mode) {
    int wid  = (blockIdx.x * blockDim.x + threadIdx.x) >> 5;
    int lane = threadIdx.x & 31;
    int mtiles = M >> 4;
    int ngroups = N >> 6;
    if (wid >= mtiles * ngroups) return;       // whole-wave uniform guard
    int m0 = (wid % mtiles) << 4;
    int n0 = (wid / mtiles) << 6;
    v8f acc[4];
#pragma unroll
    for (int j = 0; j < 4; j++) acc[j] = vzero();

    v16bf ac = load_a_bf(A, K, m0, 0, lane);
    v16bf bc[4];
#pragma unroll
    for (int j = 0; j < 4; j++) bc[j] = load_b_bf(W, K, n0 + 16 * j, 0, lane);

    for (int kk = 32; kk < K; kk += 32) {
        v16bf an = load_a_bf(A, K, m0, kk, lane);
        v16bf bn[4];
#pragma unroll
        for (int j = 0; j < 4; j++) bn[j] = load_b_bf(W, K, n0 + 16 * j, kk, lane);
#pragma unroll
        for (int j = 0; j < 4; j++) acc[j] = wmma_bf16(ac, bc[j], acc[j]);
        ac = an;
#pragma unroll
        for (int j = 0; j < 4; j++) bc[j] = bn[j];
    }
#pragma unroll
    for (int j = 0; j < 4; j++) acc[j] = wmma_bf16(ac, bc[j], acc[j]);

    int rowoff = (lane & 16) ? 8 : 0;
    int cl = lane & 15;
#pragma unroll
    for (int j = 0; j < 4; j++) {
        int col = n0 + 16 * j + cl;
        float bv = bias[col];
#pragma unroll
        for (int i = 0; i < 8; i++) {
            float v = acc[j][i] + bv;
            if (mode != 0) v = fmaxf(v, 0.f);
            size_t idx = (size_t)(m0 + i + rowoff) * (size_t)N + col;
            if (mode == 1) Cb[idx] = f2bfraw(v);
            else           Cf[idx] = v;
        }
    }
}

// ---------------- fused tri-modal LSTM step ----------------
struct LstmArgs {
    const unsigned short* x[3];     // bf16 x-slice at step t, [NB, D]
    const unsigned short* hin[3];   // bf16 [NB, H]
    unsigned short* hout[3];        // bf16 [NB, H]
    float* c[3];                    // fp32 [NB, H], in-place
    const unsigned short* wih[3];
    const unsigned short* whh[3];
    const float* bih[3];
    const float* bhh[3];
    float* cc_tt;                   // fp32 [NB, 2*HD] (for softmax product)
    unsigned short* cc_ttb;         // bf16 [NB, 2*HD] (GEMM A operand)
    unsigned short* oT;             // bf16 [NB, HD+MEM] gather
    const int* lengths;
    int t;
};

// pipelined 4-gate accumulation over one operand stream (x@Wih or h@Whh)
DEV void lstm_mm(const unsigned short* __restrict__ X, const unsigned short* __restrict__ Wt,
                 int Kdim, int H, int m0, int o0, int lane, v8f acc[4]) {
    v16bf ac = load_a_bf(X, Kdim, m0, 0, lane);
    v16bf bc[4];
#pragma unroll
    for (int g = 0; g < 4; g++) bc[g] = load_b_bf(Wt, Kdim, g * H + o0, 0, lane);
    for (int kk = 32; kk < Kdim; kk += 32) {
        v16bf an = load_a_bf(X, Kdim, m0, kk, lane);
        v16bf bn[4];
#pragma unroll
        for (int g = 0; g < 4; g++) bn[g] = load_b_bf(Wt, Kdim, g * H + o0, kk, lane);
#pragma unroll
        for (int g = 0; g < 4; g++) acc[g] = wmma_bf16(ac, bc[g], acc[g]);
        ac = an;
#pragma unroll
        for (int g = 0; g < 4; g++) bc[g] = bn[g];
    }
#pragma unroll
    for (int g = 0; g < 4; g++) acc[g] = wmma_bf16(ac, bc[g], acc[g]);
}

__global__ void k_lstm(LstmArgs A) {
    int wid  = (blockIdx.x * blockDim.x + threadIdx.x) >> 5;
    int lane = threadIdx.x & 31;
    int mod, tile;
    if (wid < 256)      { mod = 0; tile = wid; }
    else if (wid < 384) { mod = 1; tile = wid - 256; }
    else if (wid < 512) { mod = 2; tile = wid - 384; }
    else return;
    const int H = (mod == 0) ? H_T : H_A;
    const int D = (mod == 0) ? D_T : ((mod == 1) ? D_A : D_V);
    const int modoff = (mod == 0) ? 0 : ((mod == 1) ? 256 : 384);
    int m0 = (tile & 15) << 4;
    int o0 = (tile >> 4) << 4;

    v8f acc[4];
#pragma unroll
    for (int g = 0; g < 4; g++) acc[g] = vzero();

    lstm_mm(A.x[mod],   A.wih[mod], D, H, m0, o0, lane, acc);   // x @ Wih^T
    lstm_mm(A.hin[mod], A.whh[mod], H, H, m0, o0, lane, acc);   // h @ Whh^T

    int rowoff = (lane & 16) ? 8 : 0;
    int col = o0 + (lane & 15);
    float bI = A.bih[mod][0 * H + col] + A.bhh[mod][0 * H + col];
    float bF = A.bih[mod][1 * H + col] + A.bhh[mod][1 * H + col];
    float bG = A.bih[mod][2 * H + col] + A.bhh[mod][2 * H + col];
    float bO = A.bih[mod][3 * H + col] + A.bhh[mod][3 * H + col];
    float* cst  = A.c[mod];
    unsigned short* hout = A.hout[mod];
#pragma unroll
    for (int i = 0; i < 8; i++) {
        int r = m0 + i + rowoff;
        float gi = sigmoidf_(acc[0][i] + bI);
        float gf = sigmoidf_(acc[1][i] + bF);
        float gg = tanhf(acc[2][i] + bG);
        float go = sigmoidf_(acc[3][i] + bO);
        float cold = cst[(size_t)r * H + col];
        float c2 = gf * cold + gi * gg;
        float h2 = go * tanhf(c2);
        cst[(size_t)r * H + col]  = c2;
        hout[(size_t)r * H + col] = f2bfraw(h2);
        A.cc_tt [(size_t)r * 1024 + modoff + col]        = cold;            // prev_cc
        A.cc_tt [(size_t)r * 1024 + 512 + modoff + col]  = c2;              // cc
        A.cc_ttb[(size_t)r * 1024 + modoff + col]        = f2bfraw(cold);
        A.cc_ttb[(size_t)r * 1024 + 512 + modoff + col]  = f2bfraw(c2);
        if (A.lengths[r] - 1 == A.t) A.oT[(size_t)r * 768 + modoff + col] = f2bfraw(h2);
    }
}

// ---------------- row softmax + cc_star = cc_tt * attn (bf16 out) ----------------
__global__ void k_softmax(const float* __restrict__ logits, const float* __restrict__ cc_tt,
                          unsigned short* __restrict__ cc_star) {
    __shared__ float red[256];
    int r = blockIdx.x;
    int tid = threadIdx.x;
    float v[4];
    float vmax = -3.4e38f;
#pragma unroll
    for (int j = 0; j < 4; j++) {
        v[j] = logits[(size_t)r * 1024 + tid + 256 * j];
        vmax = fmaxf(vmax, v[j]);
    }
    red[tid] = vmax; __syncthreads();
    for (int s = 128; s > 0; s >>= 1) { if (tid < s) red[tid] = fmaxf(red[tid], red[tid + s]); __syncthreads(); }
    vmax = red[0]; __syncthreads();
    float ssum = 0.f;
#pragma unroll
    for (int j = 0; j < 4; j++) { v[j] = expf(v[j] - vmax); ssum += v[j]; }
    red[tid] = ssum; __syncthreads();
    for (int s = 128; s > 0; s >>= 1) { if (tid < s) red[tid] += red[tid + s]; __syncthreads(); }
    float inv = 1.f / red[0];
#pragma unroll
    for (int j = 0; j < 4; j++) {
        size_t idx = (size_t)r * 1024 + tid + 256 * j;
        cc_star[idx] = f2bfraw(cc_tt[idx] * v[j] * inv);
    }
}

// ---------------- fused g1/g2/u layer2 + memory update ----------------
__global__ void k_gmlp2(const unsigned short* __restrict__ Y2,  // bf16 [NB,1536]: g1|g2|u layer-1
                        const unsigned short* __restrict__ W1, const unsigned short* __restrict__ W2,
                        const unsigned short* __restrict__ W3,
                        const float* __restrict__ b1, const float* __restrict__ b2, const float* __restrict__ b3,
                        float* __restrict__ mem, const int* __restrict__ lengths,
                        unsigned short* __restrict__ oT, int t) {
    int wid  = (blockIdx.x * blockDim.x + threadIdx.x) >> 5;
    int lane = threadIdx.x & 31;
    if (wid >= 256) return;
    int m0 = (wid & 15) << 4;
    int o0 = (wid >> 4) << 4;
    v8f p1 = vzero(), p2 = vzero(), p3 = vzero();

    v16bf a1c = load_a_bf(Y2,        1536, m0, 0, lane);
    v16bf a2c = load_a_bf(Y2 + 512,  1536, m0, 0, lane);
    v16bf a3c = load_a_bf(Y2 + 1024, 1536, m0, 0, lane);
    v16bf w1c = load_b_bf(W1, 512, o0, 0, lane);
    v16bf w2c = load_b_bf(W2, 512, o0, 0, lane);
    v16bf w3c = load_b_bf(W3, 512, o0, 0, lane);
    for (int kk = 32; kk < 512; kk += 32) {
        v16bf a1n = load_a_bf(Y2,        1536, m0, kk, lane);
        v16bf a2n = load_a_bf(Y2 + 512,  1536, m0, kk, lane);
        v16bf a3n = load_a_bf(Y2 + 1024, 1536, m0, kk, lane);
        v16bf w1n = load_b_bf(W1, 512, o0, kk, lane);
        v16bf w2n = load_b_bf(W2, 512, o0, kk, lane);
        v16bf w3n = load_b_bf(W3, 512, o0, kk, lane);
        p1 = wmma_bf16(a1c, w1c, p1);
        p2 = wmma_bf16(a2c, w2c, p2);
        p3 = wmma_bf16(a3c, w3c, p3);
        a1c = a1n; a2c = a2n; a3c = a3n;
        w1c = w1n; w2c = w2n; w3c = w3n;
    }
    p1 = wmma_bf16(a1c, w1c, p1);
    p2 = wmma_bf16(a2c, w2c, p2);
    p3 = wmma_bf16(a3c, w3c, p3);

    int rowoff = (lane & 16) ? 8 : 0;
    int col = o0 + (lane & 15);
    float bv1 = b1[col], bv2 = b2[col], bv3 = b3[col];
#pragma unroll
    for (int i = 0; i < 8; i++) {
        int r = m0 + i + rowoff;
        float g1 = sigmoidf_(p1[i] + bv1);
        float g2 = sigmoidf_(p2[i] + bv2);
        float uu = tanhf(p3[i] + bv3);
        float m = g1 * mem[(size_t)r * MEM + col] + g2 * uu;
        mem[(size_t)r * MEM + col] = m;
        if (lengths[r] - 1 == t) oT[(size_t)r * 768 + 512 + col] = f2bfraw(m);
    }
}

// ---------------- final projection to OUT=1 ----------------
__global__ void k_out2(const float* __restrict__ Yo, const float* __restrict__ ow2,
                       const float* __restrict__ ob2, float* __restrict__ out) {
    int wid  = (blockIdx.x * blockDim.x + threadIdx.x) >> 5;
    int lane = threadIdx.x & 31;
    if (wid >= NB) return;
    float s = 0.f;
    for (int k = lane; k < 768; k += 32) s += Yo[(size_t)wid * 768 + k] * ow2[k];
#pragma unroll
    for (int m = 16; m >= 1; m >>= 1) s += __shfl_xor(s, m, 32);
    if (lane == 0) out[wid] = s + ob2[0];
}

// ---------------- host launcher ----------------
static inline size_t alignup256(size_t x) { return (x + 255) & ~(size_t)255; }

extern "C" void kernel_launch(void* const* d_in, const int* in_sizes, int n_in,
                              void* d_out, int out_size, void* d_ws, size_t ws_size,
                              hipStream_t stream) {
    (void)in_sizes; (void)n_in; (void)out_size; (void)ws_size;
    const float* x_f32[3] = { (const float*)d_in[0], (const float*)d_in[1], (const float*)d_in[2] };
    const int*   lengths  = (const int*)d_in[3];
    const float* Wih[3] = { (const float*)d_in[4],  (const float*)d_in[8],  (const float*)d_in[12] };
    const float* Whh[3] = { (const float*)d_in[5],  (const float*)d_in[9],  (const float*)d_in[13] };
    const float* bih[3] = { (const float*)d_in[6],  (const float*)d_in[10], (const float*)d_in[14] };
    const float* bhh[3] = { (const float*)d_in[7],  (const float*)d_in[11], (const float*)d_in[15] };
    const float* aw1 = (const float*)d_in[16]; const float* ab1 = (const float*)d_in[17];
    const float* aw2 = (const float*)d_in[18]; const float* ab2 = (const float*)d_in[19];
    const float* g1w1 = (const float*)d_in[20]; const float* g1b1 = (const float*)d_in[21];
    const float* g1w2 = (const float*)d_in[22]; const float* g1b2 = (const float*)d_in[23];
    const float* g2w1 = (const float*)d_in[24]; const float* g2b1 = (const float*)d_in[25];
    const float* g2w2 = (const float*)d_in[26]; const float* g2b2 = (const float*)d_in[27];
    const float* uw1  = (const float*)d_in[28]; const float* ub1  = (const float*)d_in[29];
    const float* uw2  = (const float*)d_in[30]; const float* ub2  = (const float*)d_in[31];
    const float* ow1  = (const float*)d_in[32]; const float* ob1  = (const float*)d_in[33];
    const float* ow2  = (const float*)d_in[34]; const float* ob2  = (const float*)d_in[35];

    char* ws = (char*)d_ws;
    size_t off = 0;
    auto carveU = [&](size_t n) -> unsigned short* {
        unsigned short* p = (unsigned short*)(ws + off); off = alignup256(off + n * 2); return p;
    };
    auto carveF = [&](size_t n) -> float* {
        float* p = (float*)(ws + off); off = alignup256(off + n * 4); return p;
    };

    const size_t xelems[3] = { (size_t)TT*NB*D_T, (size_t)TT*NB*D_A, (size_t)TT*NB*D_V };

    // bf16 mirrors: inputs + weights
    unsigned short* xb[3]   = { carveU(xelems[0]), carveU(xelems[1]), carveU(xelems[2]) };
    unsigned short* wihb[3] = { carveU(4*H_T*D_T), carveU(4*H_A*D_A), carveU(4*H_V*D_V) };
    unsigned short* whhb[3] = { carveU(4*H_T*H_T), carveU(4*H_A*H_A), carveU(4*H_V*H_V) };
    unsigned short* aw1b  = carveU((size_t)HD * 2*HD);
    unsigned short* aw2b  = carveU((size_t)2*HD * HD);
    unsigned short* wcat1 = carveU((size_t)3*HD * 2*HD);   // g1w1|g2w1|uw1, each [HD,2HD]
    unsigned short* g1w2b = carveU((size_t)MEM * HD);
    unsigned short* g2w2b = carveU((size_t)MEM * HD);
    unsigned short* uw2b  = carveU((size_t)MEM * HD);
    unsigned short* ow1b  = carveU((size_t)768 * 768);

    // bf16 state / activations
    unsigned short* h0[3] = { carveU(NB*H_T), carveU(NB*H_A), carveU(NB*H_V) };
    unsigned short* h1[3] = { carveU(NB*H_T), carveU(NB*H_A), carveU(NB*H_V) };
    unsigned short* cc_ttb  = carveU((size_t)NB * 2*HD);
    unsigned short* Y1      = carveU((size_t)NB * HD);
    unsigned short* cc_star = carveU((size_t)NB * 2*HD);
    unsigned short* Y2      = carveU((size_t)NB * 3*HD);
    unsigned short* oT      = carveU((size_t)NB * (HD + MEM));

    // fp32 buffers
    float* bcat1  = carveF(3 * HD);
    float* cst[3] = { carveF(NB*H_T), carveF(NB*H_A), carveF(NB*H_V) };
    float* mem    = carveF(NB * MEM);
    float* cc_tt  = carveF((size_t)NB * 2*HD);
    float* logits = carveF((size_t)NB * 2*HD);
    float* Yo     = carveF((size_t)NB * (HD + MEM));

    // ---- once-per-launch setup (recomputed every call; deterministic) ----
    auto cvt = [&](const float* src, unsigned short* dst, size_t n) {
        k_cvt<<<dim3((unsigned)((n + 255) / 256)), dim3(256), 0, stream>>>(src, dst, (int)n);
    };
    cvt(x_f32[0], xb[0], xelems[0]);
    cvt(x_f32[1], xb[1], xelems[1]);
    cvt(x_f32[2], xb[2], xelems[2]);
    cvt(Wih[0], wihb[0], (size_t)4*H_T*D_T);
    cvt(Wih[1], wihb[1], (size_t)4*H_A*D_A);
    cvt(Wih[2], wihb[2], (size_t)4*H_V*D_V);
    cvt(Whh[0], whhb[0], (size_t)4*H_T*H_T);
    cvt(Whh[1], whhb[1], (size_t)4*H_A*H_A);
    cvt(Whh[2], whhb[2], (size_t)4*H_V*H_V);
    cvt(aw1, aw1b, (size_t)HD * 2*HD);
    cvt(aw2, aw2b, (size_t)2*HD * HD);
    cvt(g1w1, wcat1,                       (size_t)HD * 2*HD);
    cvt(g2w1, wcat1 + (size_t)HD * 2*HD,   (size_t)HD * 2*HD);
    cvt(uw1,  wcat1 + (size_t)2*HD * 2*HD, (size_t)HD * 2*HD);
    cvt(g1w2, g1w2b, (size_t)MEM * HD);
    cvt(g2w2, g2w2b, (size_t)MEM * HD);
    cvt(uw2,  uw2b,  (size_t)MEM * HD);
    cvt(ow1,  ow1b,  (size_t)768 * 768);

    hipMemcpyAsync(bcat1,          g1b1, HD * sizeof(float), hipMemcpyDeviceToDevice, stream);
    hipMemcpyAsync(bcat1 + HD,     g2b1, HD * sizeof(float), hipMemcpyDeviceToDevice, stream);
    hipMemcpyAsync(bcat1 + 2 * HD, ub1,  HD * sizeof(float), hipMemcpyDeviceToDevice, stream);

    // zero recurrent state (fresh each call; bf16 0x0000 == 0.0)
    for (int m = 0; m < 3; m++) {
        int H = (m == 0) ? H_T : H_A;
        hipMemsetAsync(h0[m], 0, (size_t)NB * H * sizeof(unsigned short), stream);
        hipMemsetAsync(h1[m], 0, (size_t)NB * H * sizeof(unsigned short), stream);
        hipMemsetAsync(cst[m], 0, (size_t)NB * H * sizeof(float), stream);
    }
    hipMemsetAsync(mem, 0, (size_t)NB * MEM * sizeof(float), stream);

    auto gemm = [&](const unsigned short* A, const unsigned short* W, const float* bias,
                    float* Cf, unsigned short* Cb, int M, int N, int K, int mode) {
        int waves = (M >> 4) * (N >> 6);
        k_gemm<<<dim3((unsigned)((waves + 7) / 8)), dim3(256), 0, stream>>>(A, W, bias, Cf, Cb, M, N, K, mode);
    };

    const size_t xstride[3] = { (size_t)NB * D_T, (size_t)NB * D_A, (size_t)NB * D_V };

    LstmArgs la;
    for (int m = 0; m < 3; m++) {
        la.wih[m] = wihb[m]; la.whh[m] = whhb[m];
        la.bih[m] = bih[m];  la.bhh[m] = bhh[m];
        la.c[m] = cst[m];
    }
    la.cc_tt = cc_tt; la.cc_ttb = cc_ttb; la.oT = oT; la.lengths = lengths;

    // ---- sequential scan: 6 kernels per timestep ----
    for (int t = 0; t < TT; t++) {
        int par = t & 1;
        for (int m = 0; m < 3; m++) {
            la.x[m]    = xb[m] + (size_t)t * xstride[m];
            la.hin[m]  = par ? h1[m] : h0[m];
            la.hout[m] = par ? h0[m] : h1[m];
        }
        la.t = t;
        k_lstm<<<dim3(64), dim3(256), 0, stream>>>(la);                                    // 512 waves
        gemm(cc_ttb,  aw1b,  ab1,   nullptr, Y1,   NB, HD,   2*HD, 1);                     // attn L1 + relu -> bf16
        gemm(Y1,      aw2b,  ab2,   logits,  nullptr, NB, 2*HD, HD, 0);                    // attn L2 -> fp32
        k_softmax<<<dim3(NB), dim3(256), 0, stream>>>(logits, cc_tt, cc_star);             // softmax * cc_tt -> bf16
        gemm(cc_star, wcat1, bcat1, nullptr, Y2,   NB, 3*HD, 2*HD, 1);                     // g1|g2|u L1 + relu -> bf16
        k_gmlp2<<<dim3(32), dim3(256), 0, stream>>>(Y2, g1w2b, g2w2b, uw2b,
                                                    g1b2, g2b2, ub2, mem, lengths, oT, t);
    }

    // ---- output MLP on gathered last-step states ----
    gemm(oT, ow1b, ob1, Yo, nullptr, NB, 768, 768, 2);                                     // relu -> fp32
    k_out2<<<dim3(8), dim3(256), 0, stream>>>(Yo, ow2, ob2, (float*)d_out);
}
